// MixtureOfExperts_7808250544570
// MI455X (gfx1250) — matrix operand order repaired and checked by reading, not compiled, not run on hardware
//
#include <hip/hip_runtime.h>
#include <hip/hip_bf16.h>
#include <stdint.h>
#include <stddef.h>

#define N_TOK 16384
#define DIM   1024
#define HID   256
#define OUTD  1024
#define NEXP  8

#define XS_STRIDE (DIM + 8)   // halves; +8 keeps token rows on different bank groups
#define HS_STRIDE (HID + 8)

typedef __attribute__((ext_vector_type(16))) __bf16 v16bf;
typedef __attribute__((ext_vector_type(8)))  __bf16 v8bf;
typedef __attribute__((ext_vector_type(4)))  __bf16 v4bf;
typedef __attribute__((ext_vector_type(8)))  float  v8f;
typedef __attribute__((ext_vector_type(4)))  float  v4f;

__device__ __forceinline__ v16bf bcat(v8bf lo, v8bf hi) {
  return __builtin_shufflevector(lo, hi, 0,1,2,3,4,5,6,7,8,9,10,11,12,13,14,15);
}

__device__ __forceinline__ v8f wmma_bf16(v16bf a, v16bf b, v8f c) {
  // D = A(16x32) * B(32x16) + C, f32 accum
  return __builtin_amdgcn_wmma_f32_16x16x32_bf16(false, a, false, b, (short)0, c, false, false);
}

// A-operand (16-bit 16x32): lane owns row m; elems 0..7 -> K=kb+8h+e, elems 8..15 -> K=kb+16+8h+e
__device__ __forceinline__ v16bf load_a(const __bf16* row, int kb, int h) {
  v8bf lo = *(const v8bf*)(row + kb + 8 * h);
  v8bf hi = *(const v8bf*)(row + kb + 16 + 8 * h);
  return bcat(lo, hi);
}

// B-operand (16-bit 32x16): lane owns col n (stored as LDS row); elems 0..7 -> K=kb+8h+e, elems 8..15 -> K=kb+16+8h+e
__device__ __forceinline__ v16bf load_b(const __bf16* row, int kb, int h) {
  v8bf lo = *(const v8bf*)(row + kb + 8 * h);
  v8bf hi = *(const v8bf*)(row + kb + 16 + 8 * h);
  return bcat(lo, hi);
}

__device__ __forceinline__ float gelu_f(float v) {
  return 0.5f * v * (1.0f + erff(v * 0.70710678118654752440f));
}

// ---------------- prologue kernels ----------------

// gate: gws[n][e] = softmax(x @ Wg + bg); one wave per row
__global__ __launch_bounds__(256) void gate_kernel(const float* __restrict__ x,
                                                   const float* __restrict__ Wg,
                                                   const float* __restrict__ bg,
                                                   float* __restrict__ gws) {
  const int row  = blockIdx.x * 8 + (threadIdx.x >> 5);
  const int lane = threadIdx.x & 31;
  float acc[NEXP];
#pragma unroll
  for (int e = 0; e < NEXP; ++e) acc[e] = 0.0f;
  const float* xr = x + (size_t)row * DIM;
  for (int d = lane; d < DIM; d += 32) {
    const float xv = xr[d];
    const float* wr = Wg + (size_t)d * NEXP;
#pragma unroll
    for (int e = 0; e < NEXP; ++e) acc[e] += xv * wr[e];
  }
#pragma unroll
  for (int e = 0; e < NEXP; ++e)
#pragma unroll
    for (int off = 16; off > 0; off >>= 1) acc[e] += __shfl_xor(acc[e], off, 32);
  if (lane == 0) {
    float m = -3.4e38f;
#pragma unroll
    for (int e = 0; e < NEXP; ++e) { acc[e] += bg[e]; m = fmaxf(m, acc[e]); }
    float s = 0.0f;
#pragma unroll
    for (int e = 0; e < NEXP; ++e) { acc[e] = expf(acc[e] - m); s += acc[e]; }
    const float inv = 1.0f / s;
#pragma unroll
    for (int e = 0; e < NEXP; ++e) gws[(size_t)row * NEXP + e] = acc[e] * inv;
  }
}

// x f32 -> bf16 (4 elems / thread)
__global__ __launch_bounds__(256) void cvt_x_kernel(const float* __restrict__ x,
                                                    __bf16* __restrict__ xb) {
  const size_t i = ((size_t)blockIdx.x * 256 + threadIdx.x) * 4;
  v4f v = *(const v4f*)(x + i);
  *(v4bf*)(xb + i) = __builtin_convertvector(v, v4bf);
}

// w1t[e][hcol][d] = bf16(W1[e][d][hcol])
__global__ __launch_bounds__(256) void cvt_w1t_kernel(const float* __restrict__ W1,
                                                      __bf16* __restrict__ w1t) {
  const size_t i = (size_t)blockIdx.x * 256 + threadIdx.x;  // E*H*D
  const size_t d = i % DIM;
  const size_t t = i / DIM;
  const size_t hc = t % HID;
  const size_t e = t / HID;
  w1t[i] = (__bf16)W1[e * (size_t)DIM * HID + d * HID + hc];
}

// w2t[e][o][hc] = bf16(W2[e][hc][o])
__global__ __launch_bounds__(256) void cvt_w2t_kernel(const float* __restrict__ W2,
                                                      __bf16* __restrict__ w2t) {
  const size_t i = (size_t)blockIdx.x * 256 + threadIdx.x;  // E*O*H
  const size_t hc = i % HID;
  const size_t t = i / HID;
  const size_t o = t % OUTD;
  const size_t e = t / OUTD;
  w2t[i] = (__bf16)W2[e * (size_t)HID * OUTD + hc * OUTD + o];
}

// ---------------- fused MoE kernel ----------------
// block: 16 tokens, 256 threads (8 waves). Wave w owns H-slice [w*32,w*32+32) in
// phase A and O-slice [w*128,(w+1)*128) in phase B. Gate weight folded into h.
__global__ __launch_bounds__(256) void moe_fused_kernel(
    const __bf16* __restrict__ xb, const __bf16* __restrict__ w1t,
    const float* __restrict__ b1p, const __bf16* __restrict__ w2t,
    const float* __restrict__ b2p, const float* __restrict__ gws,
    float* __restrict__ out) {
  __shared__ __align__(32) __bf16 xs[16 * XS_STRIDE];
  __shared__ __align__(32) __bf16 hs[16 * HS_STRIDE];

  const int tid  = threadIdx.x;
  const int wave = tid >> 5;
  const int lane = tid & 31;
  const int h    = lane >> 4;   // lane half (0/1)
  const int l16  = lane & 15;
  const int n0   = blockIdx.x * 16;

  // ---- stage 16 x DIM bf16 x-tile into LDS with CDNA5 async copy ----
  {
    const __bf16* src = xb + (size_t)n0 * DIM;
#pragma unroll
    for (int it = 0; it < (16 * DIM / 8) / 256; ++it) {  // 8 chunks of 16B per thread
      const int c   = tid + it * 256;
      const int row = c >> 7;           // DIM/8 = 128 chunks per row
      const int col = (c & 127) * 8;    // halves
      const unsigned lds = (unsigned)(uintptr_t)(xs + row * XS_STRIDE + col);
      const unsigned long long g = (unsigned long long)(uintptr_t)(src + (size_t)row * DIM + col);
      asm volatile("global_load_async_to_lds_b128 %0, %1, off" :: "v"(lds), "v"(g) : "memory");
    }
    asm volatile("s_wait_asynccnt 0" ::: "memory");
  }
  __syncthreads();

  v8f oacc[8];
#pragma unroll
  for (int t = 0; t < 8; ++t) oacc[t] = {};

  const __bf16* bx = xs + l16 * XS_STRIDE;  // token row: B operand, phase A
  const __bf16* bh = hs + l16 * HS_STRIDE;  // token row: B operand, phase B

  for (int e = 0; e < NEXP; ++e) {
    // ---- phase A: hT = gw * gelu(W1^T @ x^T + b1) ; C-tile M=H rows, N=tokens ----
    v8f a0 = {}, a1 = {};
    const __bf16* ar0 = w1t + ((size_t)e * HID + wave * 32 + l16) * DIM;
    const __bf16* ar1 = ar0 + (size_t)16 * DIM;
    for (int kb = 0; kb < DIM; kb += 32) {
      v16bf bv = load_b(bx, kb, h);
      a0 = wmma_bf16(load_a(ar0, kb, h), bv, a0);
      a1 = wmma_bf16(load_a(ar1, kb, h), bv, a1);
    }
    const float gw = gws[(size_t)(n0 + l16) * NEXP + e];
    v8f b1v0 = *(const v8f*)(b1p + e * HID + wave * 32 + 8 * h);
    v8f b1v1 = *(const v8f*)(b1p + e * HID + wave * 32 + 16 + 8 * h);
#pragma unroll
    for (int r = 0; r < 8; ++r) {
      a0[r] = gw * gelu_f(a0[r] + b1v0[r]);
      a1[r] = gw * gelu_f(a1[r] + b1v1[r]);
    }
    // contiguous b128 stores: hs[token][wave*32 + 8h .. +7] (+16 for tile 1)
    *(v8bf*)(hs + l16 * HS_STRIDE + wave * 32 + 8 * h)      = __builtin_convertvector(a0, v8bf);
    *(v8bf*)(hs + l16 * HS_STRIDE + wave * 32 + 16 + 8 * h) = __builtin_convertvector(a1, v8bf);
    __syncthreads();

    // ---- phase B: out^T += W2^T @ hT, accumulated across experts in C regs ----
    const __bf16* w2e = w2t + (size_t)e * OUTD * HID;
    for (int kb = 0; kb < HID; kb += 32) {
      v16bf bv = load_b(bh, kb, h);
#pragma unroll
      for (int t = 0; t < 8; ++t) {
        const __bf16* ar = w2e + (size_t)(wave * 128 + t * 16 + l16) * HID;
        oacc[t] = wmma_bf16(load_a(ar, kb, h), bv, oacc[t]);
      }
    }
    __syncthreads();  // hs reused next expert
  }

  // ---- fold bias: out += sum_e gw[e] * b2[e][:] ; then contiguous f32 stores ----
  v8f gwv = *(const v8f*)(gws + (size_t)(n0 + l16) * NEXP);
  float* orow = out + (size_t)(n0 + l16) * OUTD + wave * 128;
#pragma unroll
  for (int t = 0; t < 8; ++t) {
#pragma unroll
    for (int e = 0; e < NEXP; ++e) {
      v8f b2v = *(const v8f*)(b2p + (size_t)e * OUTD + wave * 128 + t * 16 + 8 * h);
#pragma unroll
      for (int r = 0; r < 8; ++r) oacc[t][r] += gwv[e] * b2v[r];
    }
    *(v8f*)(orow + t * 16 + 8 * h) = oacc[t];
  }
}

// ---------------- launcher ----------------
extern "C" void kernel_launch(void* const* d_in, const int* in_sizes, int n_in,
                              void* d_out, int out_size, void* d_ws, size_t ws_size,
                              hipStream_t stream) {
  const float* x  = (const float*)d_in[0];
  const float* W1 = (const float*)d_in[1];
  const float* b1 = (const float*)d_in[2];
  const float* W2 = (const float*)d_in[3];
  const float* b2 = (const float*)d_in[4];
  const float* Wg = (const float*)d_in[5];
  const float* bg = (const float*)d_in[6];
  float* out = (float*)d_out;

  char* ws = (char*)d_ws;
  float*  gws = (float*)ws;                                  // N*E*4   = 0.5 MB
  __bf16* xb  = (__bf16*)(ws + (1ull << 20));                // N*D*2   = 32 MB
  __bf16* w1t = (__bf16*)(ws + (1ull << 20) + (32ull << 20)); // E*H*D*2 = 4 MB
  __bf16* w2t = (__bf16*)(ws + (1ull << 20) + (36ull << 20)); // E*O*H*2 = 4 MB

  gate_kernel   <<<N_TOK / 8,                    256, 0, stream>>>(x, Wg, bg, gws);
  cvt_x_kernel  <<<(N_TOK * DIM) / (256 * 4),    256, 0, stream>>>(x, xb);
  cvt_w1t_kernel<<<(NEXP * HID * DIM) / 256,     256, 0, stream>>>(W1, w1t);
  cvt_w2t_kernel<<<(NEXP * HID * OUTD) / 256,    256, 0, stream>>>(W2, w2t);
  moe_fused_kernel<<<N_TOK / 16, 256, 0, stream>>>(xb, w1t, b1, w2t, b2, gws, out);
}
